// GraphExplorer_42889543418334
// MI455X (gfx1250) — compile-verified, matrix operand order checked by reading, not run on hardware
//
#include <hip/hip_runtime.h>
#include <hip/hip_bf16.h>
#include <cstdint>
#include <cstddef>

// Problem constants (from the reference)
#define Bb    256
#define Dm    512
#define Hh    8
#define HDh   64
#define Vv    40
#define Ll    80
#define Gg    76
#define GOc   1
#define STARc 39

// CDNA5 async global->LDS path (ASYNCcnt-tracked), guarded so the file compiles
// on toolchains that don't declare the builtins (falls back to VGPR round-trip).
#if defined(__AMDGCN__) && __has_builtin(__builtin_amdgcn_global_load_async_to_lds_b128) && \
    __has_builtin(__builtin_amdgcn_s_wait_asynccnt)
#define GE_ASYNC_LDS 1
#else
#define GE_ASYNC_LDS 0
#endif

typedef __attribute__((ext_vector_type(16))) _Float16 v16h_t;
typedef __attribute__((ext_vector_type(8)))  float    v8f_t;
typedef int v4i_t __attribute__((vector_size(16)));   // 4 x i32, matches b128 builtin param

union V16HU { uint4 u[2]; v16h_t v; };

// ---------------- small device helpers ----------------

__device__ __forceinline__ unsigned ge_hash(unsigned a, unsigned b, unsigned c) {
  unsigned x = a * 0x9E3779B1u + b * 0x85EBCA77u + c * 0xC2B2AE3Du + 0x27D4EB2Fu;
  x ^= x >> 16; x *= 0x7FEB352Du; x ^= x >> 15; x *= 0x846CA68Bu; x ^= x >> 16;
  return x;
}
__device__ __forceinline__ float ge_gumbel(unsigned a, unsigned b, unsigned c) {
  float u = (ge_hash(a, b, c) + 0.5f) * (1.0f / 4294967296.0f);
  return -logf(-logf(u));
}
__device__ __forceinline__ int ge_wrap(int i, int n) { int r = i % n; return r < 0 ? r + n : r; }

// Masked Gumbel-argmax sampling over n logits (softmax domain, like reference).
__device__ __forceinline__ void ge_sample_masked(const float* lg, const bool* mask, int n,
                                                 unsigned seed, unsigned b,
                                                 int* out_idx, float* out_lp) {
  float mx = -1e30f;
  for (int v = 0; v < n; ++v) mx = fmaxf(mx, lg[v]);
  float s = 0.f;
  for (int v = 0; v < n; ++v) s += expf(lg[v] - mx);
  float pmsum = 0.f, best = -1e30f, pmbest = 0.f;
  int bi = 0;
  for (int v = 0; v < n; ++v) {
    float pm = mask[v] ? 0.f : expf(lg[v] - mx) / s;
    pmsum += pm;
    float g2 = logf(pm + 1e-20f) + ge_gumbel(seed, b, (unsigned)v);
    if (g2 > best) { best = g2; bi = v; pmbest = pm; }
  }
  *out_idx = bi;
  *out_lp  = logf(pmbest + 1e-20f) - logf(pmsum + 1e-20f);
}

// ---------------- utility kernels ----------------

__global__ __launch_bounds__(256) void ge_fill_i32(int* p, int v, int n) {
  int i = blockIdx.x * 256 + threadIdx.x; if (i < n) p[i] = v;
}
__global__ __launch_bounds__(256) void ge_fill_f32(float* p, float v, int n) {
  int i = blockIdx.x * 256 + threadIdx.x; if (i < n) p[i] = v;
}
__global__ __launch_bounds__(256) void ge_copy_i32(const int* s, int* d, int n) {
  int i = blockIdx.x * 256 + threadIdx.x; if (i < n) d[i] = s[i];
}

// W f32 [K x N] row-major -> Wt f16 [Npad x K] row-major (transposed, zero padded)
__global__ __launch_bounds__(256) void ge_transpose_cast(const float* __restrict__ W,
                                                         _Float16* __restrict__ Wt,
                                                         int K, int N, int Npad) {
  int idx = blockIdx.x * 256 + threadIdx.x;
  if (idx >= Npad * K) return;
  int n = idx / K, k = idx - n * K;
  float v = (n < N) ? W[(size_t)k * N + n] : 0.f;
  Wt[idx] = (_Float16)v;
}

// ---------------- WMMA GEMM: C[256 x N] = A[256 x 512] * Bt^T + bias ----------------
// A: f16 row-major, 16-row tile staged in LDS per block (async DMA when available);
// Bt: f16 [Npad x 512] (W transposed, N-major -> contiguous b128 B fragments).
// 8 wave32 waves per block, one 16x16 v_wmma_f32_16x16x32_f16 tile per wave, 16 K-chunks.
__global__ __launch_bounds__(256) void ge_gemm_wmma(const _Float16* __restrict__ A,
                                                    const _Float16* __restrict__ Bt,
                                                    const float* __restrict__ bias,
                                                    float* __restrict__ C,
                                                    _Float16* __restrict__ C16,
                                                    int Npad, int Nvalid, int ldc) {
  __shared__ _Float16 As[16 * 512];  // 16 KB A tile, shared by 8 waves
  int mtile = blockIdx.y;

#if GE_ASYNC_LDS
  {
    // GLOBAL_LOAD_ASYNC_TO_LDS_B128: memory -> LDS without a VGPR round-trip.
    // 256 lanes x 16B per issue = 4KB/sweep; 4 sweeps for the 16KB tile.
    __attribute__((address_space(1))) v4i_t* gsrc =
        (__attribute__((address_space(1))) v4i_t*)(A + (size_t)mtile * 16 * 512);
    __attribute__((address_space(3))) v4i_t* ldst =
        (__attribute__((address_space(3))) v4i_t*)As;
#pragma unroll
    for (int u = 0; u < 4; ++u) {
      int o = threadIdx.x + u * 256;  // in 16-byte units
      __builtin_amdgcn_global_load_async_to_lds_b128(gsrc + o, ldst + o, 0, 0);
    }
    __builtin_amdgcn_s_wait_asynccnt(0);
  }
#else
  {
    const uint4* gA = (const uint4*)(A + (size_t)mtile * 16 * 512);
    uint4* sA = (uint4*)As;
    for (int u = threadIdx.x; u < 1024; u += 256) sA[u] = gA[u];
  }
#endif
  __syncthreads();

  int wave  = threadIdx.x >> 5;
  int lane  = threadIdx.x & 31;
  int ntile = blockIdx.x * 8 + wave;
  if (ntile * 16 >= Npad) return;  // uniform per wave: EXEC stays all-ones for WMMA

  int mrow = lane & 15;
  int hiK  = (lane >> 4) ? 8 : 0;          // lanes 16-31 carry the K+8..15 / K+24..31 halves
  int ncol = ntile * 16 + (lane & 15);
  const _Float16* brow = Bt + (size_t)ncol * 512;

  v8f_t acc = {};
#pragma unroll 4
  for (int kc = 0; kc < 16; ++kc) {
    int kb = kc * 32 + hiK;
    V16HU af, bf;
    af.u[0] = *(const uint4*)(&As[mrow * 512 + kb]);
    af.u[1] = *(const uint4*)(&As[mrow * 512 + kb + 16]);
    bf.u[0] = *(const uint4*)(brow + kb);
    bf.u[1] = *(const uint4*)(brow + kb + 16);
    if (kc + 1 < 16) __builtin_prefetch(brow + (kc + 1) * 32 + hiK, 0, 1);
    acc = __builtin_amdgcn_wmma_f32_16x16x32_f16(false, af.v, false, bf.v,
                                                 (short)0, acc, false, false);
  }

  if (ncol < Nvalid) {
    float bv = bias ? bias[ncol] : 0.f;
    int mbase = mtile * 16 + ((lane >> 4) ? 8 : 0);
#pragma unroll
    for (int r = 0; r < 8; ++r) {
      float val = acc[r] + bv;
      size_t o = (size_t)(mbase + r) * ldc + ncol;
      C[o] = val;
      if (C16) C16[o] = (_Float16)val;
    }
  }
}

// ---------------- embedding / cache kernels ----------------

__global__ __launch_bounds__(256) void ge_embed_pos(const int* __restrict__ srcw,
                                                    const float* __restrict__ emb_word,
                                                    const float* __restrict__ emb_site,
                                                    _Float16* __restrict__ x16, int pos) {
  int idx = blockIdx.x * 256 + threadIdx.x;
  if (idx >= Bb * Dm) return;
  int b = idx >> 9, d = idx & 511;
  const int* row = srcw + (size_t)(b * Ll + pos) * 5;
  int wi = ge_wrap(row[3] + row[0] * 4, Vv * 4);
  int si = ge_wrap(row[1] * Gg + row[2], Gg * Gg);
  x16[idx] = (_Float16)(emb_word[(size_t)wi * Dm + d] + emb_site[(size_t)si * Dm + d]);
}

__global__ __launch_bounds__(256) void ge_embed_atom(const int* __restrict__ srcw,
                                                     const float* __restrict__ emb_atom,
                                                     _Float16* __restrict__ x16, int step) {
  int idx = blockIdx.x * 256 + threadIdx.x;
  if (idx >= Bb * Dm) return;
  int b = idx >> 9, d = idx & 511;
  int a = ge_wrap(srcw[(size_t)(b * Ll + step) * 5 + 0], Vv);
  x16[idx] = (_Float16)emb_atom[(size_t)a * Dm + d];
}

__global__ __launch_bounds__(256) void ge_embed_word2(const int* __restrict__ srcw,
                                                      const float* __restrict__ emb_word,
                                                      _Float16* __restrict__ x16, int step) {
  int idx = blockIdx.x * 256 + threadIdx.x;
  if (idx >= Bb * Dm) return;
  int b = idx >> 9, d = idx & 511;
  const int* srow = srcw + (size_t)(b * Ll + step) * 5;
  int w = ge_wrap(srow[0] * 4 + srow[3], Vv * 4);
  x16[idx] = (_Float16)emb_word[(size_t)w * Dm + d];
}

__global__ __launch_bounds__(256) void ge_embed_const(const float* __restrict__ emb_word,
                                                      _Float16* __restrict__ x16) {
  int idx = blockIdx.x * 256 + threadIdx.x;
  if (idx >= Bb * Dm) return;
  int d = idx & 511;
  x16[idx] = (_Float16)emb_word[(size_t)(STARc * 4 + 1) * Dm + d];
}

__global__ __launch_bounds__(256) void ge_embed_loci(const float* __restrict__ emb_loci,
                                                     const int* __restrict__ prev_a,
                                                     _Float16* __restrict__ x16) {
  int idx = blockIdx.x * 256 + threadIdx.x;
  if (idx >= Bb * Dm) return;
  int b = idx >> 9, d = idx & 511;
  int p = ge_wrap(prev_a[b], Gg);
  x16[idx] = (_Float16)emb_loci[(size_t)p * Dm + d];
}

__global__ __launch_bounds__(256) void ge_split_kv(const float* __restrict__ qkv,
                                                   _Float16* __restrict__ Kc,
                                                   _Float16* __restrict__ Vc, int pos) {
  int idx = blockIdx.x * 256 + threadIdx.x;
  if (idx >= Bb * Dm) return;
  int b = idx >> 9, d = idx & 511;
  size_t o = (size_t)(b * Ll + pos) * Dm + d;
  Kc[o] = (_Float16)qkv[(size_t)b * 1536 + 512 + d];
  Vc[o] = (_Float16)qkv[(size_t)b * 1536 + 1024 + d];
}

// ---------------- incremental attention: one wave per (b,h) ----------------

__global__ __launch_bounds__(32) void ge_attn(const float* __restrict__ qkv,
                                              const _Float16* __restrict__ Kc,
                                              const _Float16* __restrict__ Vc,
                                              _Float16* __restrict__ attn16, int ln) {
  __shared__ float sc[Ll];
  int b = blockIdx.x >> 3;
  int h = blockIdx.x & 7;
  int lane = threadIdx.x;
  const float* q = qkv + (size_t)b * 1536 + h * HDh;
  float q0 = q[lane * 2], q1 = q[lane * 2 + 1];
  for (int j = 0; j < ln; ++j) {
    const _Float16* kr = Kc + (size_t)(b * Ll + j) * Dm + h * HDh;
    float p = q0 * (float)kr[lane * 2] + q1 * (float)kr[lane * 2 + 1];
#pragma unroll
    for (int off = 16; off > 0; off >>= 1) p += __shfl_down(p, off, 32);
    if (lane == 0) sc[j] = p * 0.125f;  // 1/sqrt(64)
  }
  __syncthreads();
  float mx = -1e30f;
  for (int j = lane; j < ln; j += 32) mx = fmaxf(mx, sc[j]);
#pragma unroll
  for (int off = 16; off > 0; off >>= 1) mx = fmaxf(mx, __shfl_xor(mx, off, 32));
  float sum = 0.f;
  for (int j = lane; j < ln; j += 32) { float e = expf(sc[j] - mx); sc[j] = e; sum += e; }
#pragma unroll
  for (int off = 16; off > 0; off >>= 1) sum += __shfl_xor(sum, off, 32);
  __syncthreads();
  float inv = 1.0f / sum;
  float o0 = 0.f, o1 = 0.f;
  for (int j = 0; j < ln; ++j) {
    float wgt = sc[j] * inv;
    const _Float16* vr = Vc + (size_t)(b * Ll + j) * Dm + h * HDh;
    o0 += wgt * (float)vr[lane * 2];
    o1 += wgt * (float)vr[lane * 2 + 1];
  }
  size_t od = (size_t)b * Dm + h * HDh + lane * 2;
  attn16[od] = (_Float16)o0;
  attn16[od + 1] = (_Float16)o1;
}

// ---------------- GRU gate fusion ----------------

__global__ __launch_bounds__(256) void ge_gru(const float* __restrict__ gi,
                                              const float* __restrict__ gh,
                                              const float* __restrict__ hprev,
                                              float* __restrict__ hout,
                                              _Float16* __restrict__ hout16) {
  int idx = blockIdx.x * 256 + threadIdx.x;
  if (idx >= Bb * Dm) return;
  int b = idx >> 9, d = idx & 511;
  size_t gb = (size_t)b * 1536;
  float ir = gi[gb + d], iz = gi[gb + 512 + d], inn = gi[gb + 1024 + d];
  float hr = gh[gb + d], hz = gh[gb + 512 + d], hn  = gh[gb + 1024 + d];
  float r = 1.f / (1.f + expf(-(ir + hr)));
  float z = 1.f / (1.f + expf(-(iz + hz)));
  float n = tanhf(inn + r * hn);
  float o = (1.f - z) * n + z * hprev[idx];
  hout[idx] = o;
  hout16[idx] = (_Float16)o;
}

// ---------------- per-sample scalar bookkeeping (one thread per b) ----------------

__global__ __launch_bounds__(256) void ge_sample_atom(const float* __restrict__ logits,
    const int* __restrict__ voc_mask, int* srcw, const int* exists, int* vals_max,
    int* curr_a, int* is_end_a, float* ll, int step, unsigned seed) {
  int b = threadIdx.x; if (blockIdx.x || b >= Bb) return;
  const int* exb = exists + (size_t)b * Gg * Gg;
  int* vm = vals_max + (size_t)b * Gg;
  int curr = curr_a[b], is_end = is_end_a[b];
  int* srow = srcw + (size_t)(b * Ll + step) * 5;
  int grow = srow[4] == 0;
  int cwi = ge_wrap(curr, Gg);
  int colsum[Gg];
  for (int g = 0; g < Gg; ++g) colsum[g] = 0;
  for (int j = 0; j < Gg; ++j) { const int* er = exb + j * Gg; for (int g = 0; g < Gg; ++g) colsum[g] += er[g]; }

  bool mask[Vv];
  for (int v = 0; v < Vv; ++v) mask[v] = voc_mask[v] < 0;
  bool allj = true;
  for (int g = 0; g < Gg; ++g) {
    int vr = vm[g] - colsum[g];
    bool j = (vr == 0) || (exb[cwi * Gg + g] != 0);
    if (g == cwi) j = true;
    allj = allj && j;
  }
  int vr_curr = vm[cwi] - colsum[cwi];
  bool last = allj || (vr_curr == 0);
  if (step <= 2) last = true;
  mask[Vv - 1] = mask[Vv - 1] || last;
  mask[1] = true;
  if (is_end) for (int v = 1; v < Vv; ++v) mask[v] = true;

  int bi; float lp;
  ge_sample_masked(logits + (size_t)b * Vv, mask, Vv, seed, (unsigned)b, &bi, &lp);

  int atom = grow ? bi : srow[0];
  srow[0] = atom;
  if (grow && !is_end) ll[b] += lp;
  if ((atom == 0) && grow) is_end = 1;
  int num = 0;
  for (int g = 0; g < Gg; ++g) if (vm[g] > 0) num++;
  if (num > Gg - 1) num = Gg - 1;
  vm[num] = voc_mask[ge_wrap(atom, Vv)];
  int bud = (atom != STARc) ? 1 : 0;
  curr = is_end ? 0 : curr + bud;
  curr_a[b] = curr; is_end_a[b] = is_end;
  srow[1] = curr;
}

__global__ __launch_bounds__(256) void ge_sample_bond(const float* __restrict__ logits,
    int* srcw, const int* exists, const int* vals_max, const int* curr_a,
    const int* is_end_a, float* ll, int step, unsigned seed) {
  int b = threadIdx.x; if (blockIdx.x || b >= Bb) return;
  const int* exb = exists + (size_t)b * Gg * Gg;
  const int* vm = vals_max + (size_t)b * Gg;
  int curr = curr_a[b], is_end = is_end_a[b];
  int* srow = srcw + (size_t)(b * Ll + step) * 5;
  int grow = srow[4] == 0;
  int cwi = ge_wrap(curr, Gg);
  int colsum[Gg];
  for (int g = 0; g < Gg; ++g) colsum[g] = 0;
  for (int j = 0; j < Gg; ++j) { const int* er = exb + j * Gg; for (int g = 0; g < Gg; ++g) colsum[g] += er[g]; }
  int vr_curr = vm[cwi] - colsum[cwi];
  bool bm[4];
  bm[0] = is_end ? false : (step != 1);
  for (int i = 1; i < 4; ++i) {
    bool allj = true;
    for (int g = 0; g < Gg; ++g) {
      int vr = vm[g] - colsum[g];
      bool j = (vr < i) || (exb[cwi * Gg + g] != 0);
      if (g == cwi) j = true;
      allj = allj && j;
    }
    bm[i] = allj || (vr_curr < i) || (is_end != 0);
  }
  int bi; float lp;
  ge_sample_masked(logits + (size_t)b * 4, bm, 4, seed, (unsigned)b, &bi, &lp);
  int bond = grow ? bi : srow[3];
  srow[3] = bond;
  if (grow && !is_end) ll[b] += lp;
}

__global__ __launch_bounds__(256) void ge_sample_prev(const float* __restrict__ logits,
    int* srcw, int* exists, const int* vals_max, int* frg, const int* curr_a,
    int* is_end_a, float* ll, int step, unsigned seed) {
  int b = threadIdx.x; if (blockIdx.x || b >= Bb) return;
  int* exb = exists + (size_t)b * Gg * Gg;
  const int* vm = vals_max + (size_t)b * Gg;
  int* frgb = frg + (size_t)b * Gg;
  int curr = curr_a[b], is_end = is_end_a[b];
  int* srow = srcw + (size_t)(b * Ll + step) * 5;
  int grow = srow[4] == 0;
  int bond = srow[3];
  int cwi = ge_wrap(curr, Gg);
  int colsum[Gg];
  for (int g = 0; g < Gg; ++g) colsum[g] = 0;
  for (int j = 0; j < Gg; ++j) { const int* er = exb + j * Gg; for (int g = 0; g < Gg; ++g) colsum[g] += er[g]; }

  bool pmask[Gg];
  for (int g = 0; g < Gg; ++g)
    pmask[g] = (vm[g] == 0) || (exb[cwi * Gg + g] != 0) || ((vm[g] - colsum[g]) < bond);
  pmask[cwi] = true;
  bool col0 = (step <= 2) ? false : pmask[0];
  pmask[0] = is_end ? false : col0;
  if (is_end) for (int g = 1; g < Gg; ++g) pmask[g] = true;

  int bi; float lp;
  ge_sample_masked(logits + (size_t)b * Gg, pmask, Gg, seed, (unsigned)b, &bi, &lp);
  int prev = grow ? bi : srow[2];
  srow[2] = prev;
  if (grow && !is_end) ll[b] += lp;

  int pwi = ge_wrap(prev, Gg);
  int bud = (srow[0] != STARc) ? 1 : 0;
  int at_curr = (!grow) ? srow[4] : (bud ? frgb[pwi] : frgb[cwi]);
  frgb[cwi] = at_curr;
  int obj = frgb[cwi], tgt = frgb[pwi];
  for (int g = 0; g < Gg; ++g) if (frgb[g] == tgt) frgb[g] = obj;
  exb[cwi * Gg + pwi] = bond;
  exb[pwi * Gg + cwi] = bond;

  bool allz = true;
  for (int g = 0; g < Gg; ++g) {
    int cs = 0;
    for (int j = 0; j < Gg; ++j) cs += exb[j * Gg + g];
    if (vm[g] - cs != 0) allz = false;
  }
  if (allz) is_end = 1;
  is_end_a[b] = is_end;
}

__global__ __launch_bounds__(256) void ge_sample2a(const float* __restrict__ logits,
    const int* exists, const int* vals_max, const int* frg, int* vr_ws, int* prev_a,
    int* is_end_a, float* ll, unsigned seed) {
  int b = threadIdx.x; if (blockIdx.x || b >= Bb) return;
  const int* exb = exists + (size_t)b * Gg * Gg;
  const int* vm = vals_max + (size_t)b * Gg;
  const int* frgb = frg + (size_t)b * Gg;
  int is_end = is_end_a[b];
  int vr[Gg];
  for (int g = 0; g < Gg; ++g) vr[g] = 0;
  for (int j = 0; j < Gg; ++j) { const int* er = exb + j * Gg; for (int g = 0; g < Gg; ++g) vr[g] += er[g]; }
  for (int g = 0; g < Gg; ++g) vr[g] = vm[g] - vr[g];

  int frgs[8];
  for (int f = 0; f < 8; ++f) frgs[f] = 0;
  for (int g = 0; g < Gg; ++g) { int f = frgb[g]; if (f >= 0 && f < 8) frgs[f] += vr[g]; }
  frgs[0] = 0;
  bool allz = true;
  for (int g = 0; g < Gg; ++g) if (vr[g] != 0) allz = false;
  if (allz) is_end = 1;
  int cnt = 0;
  for (int f = 0; f < 8; ++f) if (frgs[f] != 0) cnt++;
  if (cnt <= 1) is_end = 1;

  bool mask[Gg];
  for (int g = 0; g < Gg; ++g) mask[g] = (vr[g] < 1) || (vm[g] == 0);
  if (is_end) mask[0] = false;

  int bi; float lp;
  ge_sample_masked(logits + (size_t)b * Gg, mask, Gg, seed, (unsigned)b, &bi, &lp);
  prev_a[b] = bi;
  if (!is_end) ll[b] += lp;
  is_end_a[b] = is_end;
  for (int g = 0; g < Gg; ++g) vr_ws[(size_t)b * Gg + g] = vr[g];
}

__global__ __launch_bounds__(256) void ge_sample2b(const float* __restrict__ logits,
    int* srcw, int* exists, const int* vals_max, int* frg, const int* vr_ws,
    const int* prev_a, const int* is_end_a, float* ll, int step, unsigned seed) {
  int b = threadIdx.x; if (blockIdx.x || b >= Bb) return;
  int* exb = exists + (size_t)b * Gg * Gg;
  const int* vm = vals_max + (size_t)b * Gg;
  int* frgb = frg + (size_t)b * Gg;
  const int* vr = vr_ws + (size_t)b * Gg;
  int is_end = is_end_a[b];
  int prev = prev_a[b];
  int pwi = ge_wrap(prev, Gg);

  bool mask2[Gg];
  for (int g = 0; g < Gg; ++g)
    mask2[g] = (vr[g] < 1) || (exb[pwi * Gg + g] != 0) || (vm[g] == 0) || (frgb[g] == frgb[pwi]);
  if (is_end) mask2[0] = false;

  int curr_s; float lp;
  ge_sample_masked(logits + (size_t)b * Gg, mask2, Gg, seed, (unsigned)b, &curr_s, &lp);
  if (!is_end) ll[b] += lp;

  int* srow = srcw + (size_t)(b * Ll + step) * 5;
  int r0 = STARc, r1 = curr_s, r2 = prev, r3 = 1, r4 = srow[4];
  if (is_end) { r0 = r1 = r2 = r3 = r4 = 0; }
  srow[0] = r0; srow[1] = r1; srow[2] = r2; srow[3] = r3; srow[4] = r4;

  int obj = frgb[ge_wrap(curr_s, Gg)], tgt = frgb[pwi];
  for (int g = 0; g < Gg; ++g) if (frgb[g] == tgt) frgb[g] = obj;
  int c2 = ge_wrap(srow[1], Gg), p2 = ge_wrap(srow[2], Gg), b2 = srow[3];
  exb[c2 * Gg + p2] = b2;
  exb[p2 * Gg + c2] = b2;
}

__global__ __launch_bounds__(256) void ge_set_gap(int* srcw) {
  int b = threadIdx.x; if (blockIdx.x || b >= Bb) return;
  int* srow = srcw + (size_t)(b * Ll + Gg) * 5;
  srow[0] = GOc; srow[1] = 0; srow[2] = 0; srow[3] = 0; srow[4] = 0;
}

// ---------------- host driver ----------------

extern "C" void kernel_launch(void* const* d_in, const int* in_sizes, int n_in,
                              void* d_out, int out_size, void* d_ws, size_t ws_size,
                              hipStream_t stream) {
  (void)in_sizes; (void)out_size; (void)ws_size;
  if (n_in < 20) return;

  // Inputs in setup_inputs() dict insertion order.
  const int*   src       = (const int*)d_in[0];
  const int*   voc_mask  = (const int*)d_in[1];
  const float* emb_word  = (const float*)d_in[2];
  const float* emb_site  = (const float*)d_in[3];
  const float* emb_atom  = (const float*)d_in[4];
  const float* emb_loci  = (const float*)d_in[5];
  const float* attn_Wqkv = (const float*)d_in[6];
  const float* attn_bqkv = (const float*)d_in[7];
  const float* attn_Wo   = (const float*)d_in[8];
  const float* attn_bo   = (const float*)d_in[9];
  const float* rnn_Wi    = (const float*)d_in[10];
  const float* rnn_bi    = (const float*)d_in[11];
  const float* rnn_Wh    = (const float*)d_in[12];
  const float* rnn_bh    = (const float*)d_in[13];
  const float* prj_atom_W = (const float*)d_in[14];
  const float* prj_atom_b = (const float*)d_in[15];
  const float* prj_bond_W = (const float*)d_in[16];
  const float* prj_bond_b = (const float*)d_in[17];
  const float* prj_loci_W = (const float*)d_in[18];
  const float* prj_loci_b = (const float*)d_in[19];

  // Bump allocator over workspace (aligned to 256B).
  char* wsb = (char*)d_ws; size_t off = 0;
  auto alloc = [&](size_t bytes) -> void* {
    off = (off + 255) & ~(size_t)255;
    void* p = wsb + off; off += bytes; return p;
  };

  _Float16* WqkvT  = (_Float16*)alloc((size_t)1536 * 512 * 2);
  _Float16* WoT    = (_Float16*)alloc((size_t)512 * 512 * 2);
  _Float16* WiT    = (_Float16*)alloc((size_t)1536 * 512 * 2);
  _Float16* WhT    = (_Float16*)alloc((size_t)1536 * 512 * 2);
  _Float16* WatomT = (_Float16*)alloc((size_t)48 * 512 * 2);
  _Float16* WbondT = (_Float16*)alloc((size_t)16 * 512 * 2);
  _Float16* WlociT = (_Float16*)alloc((size_t)80 * 512 * 2);
  _Float16* Kc     = (_Float16*)alloc((size_t)Bb * Ll * Dm * 2);
  _Float16* Vc     = (_Float16*)alloc((size_t)Bb * Ll * Dm * 2);
  int*      srcw   = (int*)alloc((size_t)Bb * Ll * 5 * 4);
  _Float16* x16    = (_Float16*)alloc((size_t)Bb * Dm * 2);
  float*    qkv    = (float*)alloc((size_t)Bb * 1536 * 4);
  _Float16* attn16 = (_Float16*)alloc((size_t)Bb * Dm * 2);
  float*    decA   = (float*)alloc((size_t)Bb * Dm * 4);
  _Float16* decA16 = (_Float16*)alloc((size_t)Bb * Dm * 2);
  float*    decB   = (float*)alloc((size_t)Bb * Dm * 4);
  _Float16* decB16 = (_Float16*)alloc((size_t)Bb * Dm * 2);
  float*    decL   = (float*)alloc((size_t)Bb * Dm * 4);
  _Float16* decL16 = (_Float16*)alloc((size_t)Bb * Dm * 2);
  float*    gi     = (float*)alloc((size_t)Bb * 1536 * 4);
  float*    gh     = (float*)alloc((size_t)Bb * 1536 * 4);
  float*    logits = (float*)alloc((size_t)Bb * 80 * 4);
  int*      exists = (int*)alloc((size_t)Bb * Gg * Gg * 4);
  int*      vmax   = (int*)alloc((size_t)Bb * Gg * 4);
  int*      frg    = (int*)alloc((size_t)Bb * Gg * 4);
  int*      vr_ws  = (int*)alloc((size_t)Bb * Gg * 4);
  int*      curr   = (int*)alloc((size_t)Bb * 4);
  int*      is_end = (int*)alloc((size_t)Bb * 4);
  int*      prev_a = (int*)alloc((size_t)Bb * 4);
  float*    llb    = (float*)alloc((size_t)Bb * 4);

  auto blk = [](int n) { return dim3((n + 255) / 256); };
  auto GEMM = [&](const _Float16* Aa, const _Float16* Bt, const float* bias,
                  float* C, _Float16* C16, int Npad, int Nvalid, int ldc) {
    dim3 grid((Npad / 16 + 7) / 8, 16);
    ge_gemm_wmma<<<grid, 256, 0, stream>>>(Aa, Bt, bias, C, C16, Npad, Nvalid, ldc);
  };

  // ---- prologue: f16 transposed weights + state init ----
  ge_transpose_cast<<<blk(1536 * 512), 256, 0, stream>>>(attn_Wqkv, WqkvT, 512, 1536, 1536);
  ge_transpose_cast<<<blk(512 * 512),  256, 0, stream>>>(attn_Wo,   WoT,   512, 512,  512);
  ge_transpose_cast<<<blk(1536 * 512), 256, 0, stream>>>(rnn_Wi,    WiT,   512, 1536, 1536);
  ge_transpose_cast<<<blk(1536 * 512), 256, 0, stream>>>(rnn_Wh,    WhT,   512, 1536, 1536);
  ge_transpose_cast<<<blk(48 * 512),   256, 0, stream>>>(prj_atom_W, WatomT, 512, Vv, 48);
  ge_transpose_cast<<<blk(16 * 512),   256, 0, stream>>>(prj_bond_W, WbondT, 512, 4, 16);
  ge_transpose_cast<<<blk(80 * 512),   256, 0, stream>>>(prj_loci_W, WlociT, 512, Gg, 80);

  ge_copy_i32<<<blk(Bb * Ll * 5), 256, 0, stream>>>(src, srcw, Bb * Ll * 5);
  ge_fill_i32<<<blk(Bb * Gg * Gg), 256, 0, stream>>>(exists, 0, Bb * Gg * Gg);
  ge_fill_i32<<<blk(Bb * Gg), 256, 0, stream>>>(vmax, 0, Bb * Gg);
  ge_fill_i32<<<blk(Bb * Gg), 256, 0, stream>>>(frg, 0, Bb * Gg);
  ge_fill_i32<<<blk(Bb), 256, 0, stream>>>(curr, -1, Bb);
  ge_fill_i32<<<blk(Bb), 256, 0, stream>>>(is_end, 0, Bb);
  ge_fill_f32<<<blk(Bb), 256, 0, stream>>>(llb, 0.f, Bb);

  const dim3 gBD = blk(Bb * Dm);

  // ---- phase 1: grow steps 1..G-1 ----
  for (int step = 1; step < Gg; ++step) {
    unsigned sd = 0x1000u + (unsigned)step * 7919u;
    ge_embed_pos<<<gBD, 256, 0, stream>>>(srcw, emb_word, emb_site, x16, step - 1);
    GEMM(x16, WqkvT, attn_bqkv, qkv, nullptr, 1536, 1536, 1536);
    ge_split_kv<<<gBD, 256, 0, stream>>>(qkv, Kc, Vc, step - 1);
    ge_attn<<<Bb * Hh, 32, 0, stream>>>(qkv, Kc, Vc, attn16, step);
    GEMM(attn16, WoT, attn_bo, decA, decA16, 512, 512, 512);

    GEMM(decA16, WatomT, prj_atom_b, logits, nullptr, 48, Vv, Vv);
    ge_sample_atom<<<1, 256, 0, stream>>>(logits, voc_mask, srcw, exists, vmax,
                                          curr, is_end, llb, step, sd + 1);

    ge_embed_atom<<<gBD, 256, 0, stream>>>(srcw, emb_atom, x16, step);
    GEMM(x16, WiT, rnn_bi, gi, nullptr, 1536, 1536, 1536);
    GEMM(decA16, WhT, rnn_bh, gh, nullptr, 1536, 1536, 1536);
    ge_gru<<<gBD, 256, 0, stream>>>(gi, gh, decA, decB, decB16);

    GEMM(decB16, WbondT, prj_bond_b, logits, nullptr, 16, 4, 4);
    ge_sample_bond<<<1, 256, 0, stream>>>(logits, srcw, exists, vmax, curr,
                                          is_end, llb, step, sd + 2);

    ge_embed_word2<<<gBD, 256, 0, stream>>>(srcw, emb_word, x16, step);
    GEMM(x16, WiT, rnn_bi, gi, nullptr, 1536, 1536, 1536);
    GEMM(decB16, WhT, rnn_bh, gh, nullptr, 1536, 1536, 1536);
    ge_gru<<<gBD, 256, 0, stream>>>(gi, gh, decB, decA, decA16);

    GEMM(decA16, WlociT, prj_loci_b, logits, nullptr, 80, Gg, Gg);
    ge_sample_prev<<<1, 256, 0, stream>>>(logits, srcw, exists, vmax, frg, curr,
                                          is_end, llb, step, sd + 3);
  }

  // ---- phase 2: star linking steps G+1..L-1 ----
  ge_set_gap<<<1, 256, 0, stream>>>(srcw);
  ge_fill_i32<<<blk(Bb), 256, 0, stream>>>(is_end, 0, Bb);

  for (int step = Gg + 1; step < Ll; ++step) {
    unsigned sd = 0x9000u + (unsigned)step * 7919u;
    ge_embed_pos<<<gBD, 256, 0, stream>>>(srcw, emb_word, emb_site, x16, step - 1);
    GEMM(x16, WqkvT, attn_bqkv, qkv, nullptr, 1536, 1536, 1536);
    ge_split_kv<<<gBD, 256, 0, stream>>>(qkv, Kc, Vc, step - 1);
    ge_attn<<<Bb * Hh, 32, 0, stream>>>(qkv, Kc, Vc, attn16, step);
    GEMM(attn16, WoT, attn_bo, decL, decL16, 512, 512, 512);

    ge_embed_const<<<gBD, 256, 0, stream>>>(emb_word, x16);
    GEMM(x16, WiT, rnn_bi, gi, nullptr, 1536, 1536, 1536);
    GEMM(decL16, WhT, rnn_bh, gh, nullptr, 1536, 1536, 1536);
    ge_gru<<<gBD, 256, 0, stream>>>(gi, gh, decL, decA, decA16);

    GEMM(decA16, WlociT, prj_loci_b, logits, nullptr, 80, Gg, Gg);
    ge_sample2a<<<1, 256, 0, stream>>>(logits, exists, vmax, frg, vr_ws, prev_a,
                                       is_end, llb, sd + 1);

    ge_embed_loci<<<gBD, 256, 0, stream>>>(emb_loci, prev_a, x16);
    GEMM(x16, WiT, rnn_bi, gi, nullptr, 1536, 1536, 1536);
    GEMM(decA16, WhT, rnn_bh, gh, nullptr, 1536, 1536, 1536);
    ge_gru<<<gBD, 256, 0, stream>>>(gi, gh, decA, decB, decB16);

    GEMM(decB16, WlociT, prj_loci_b, logits, nullptr, 80, Gg, Gg);
    ge_sample2b<<<1, 256, 0, stream>>>(logits, srcw, exists, vmax, frg, vr_ws,
                                       prev_a, is_end, llb, step, sd + 2);
  }

  // ---- epilogue: emit int32 src ----
  ge_copy_i32<<<blk(Bb * Ll * 5), 256, 0, stream>>>(srcw, (int*)d_out, Bb * Ll * 5);
}
